// ValkyrieLongformerAttention_35381940584739
// MI455X (gfx1250) — matrix operand order repaired and checked by reading, not compiled, not run on hardware
//
#include <hip/hip_runtime.h>
#include <hip/hip_bf16.h>

// ---------------------------------------------------------------------------
// Valkyrie Longformer attention for MI455X (gfx1250), wave32 + WMMA bf16.
//   B=1, S=2048, D=1024, H=16, HD=64, RADIUS=256, GLOBAL_IDX={0,1,2,3}
// All heavy matmuls go through v_wmma_f32_16x16x32_bf16 (f32 accumulate).
// GEMM staging uses double-buffered LDS; async memory->LDS path when the
// toolchain declares the gfx1250 async builtins (ASYNCcnt-tracked).
// ---------------------------------------------------------------------------

typedef __bf16 bf16;
typedef __attribute__((ext_vector_type(16))) bf16  v16bf;
typedef __attribute__((ext_vector_type(8)))  bf16  v8bf;
typedef __attribute__((ext_vector_type(8)))  float v8f;
typedef __attribute__((ext_vector_type(4)))  int   v4i;

#define SEQ     2048
#define DMODEL  1024
#define NHEAD   16
#define HDIM    64
#define RADIUS  256

#ifndef __has_builtin
#define __has_builtin(x) 0
#endif
#if __has_builtin(__builtin_amdgcn_global_load_async_to_lds_b128)
#define USE_ASYNC_LDS 1
#else
#define USE_ASYNC_LDS 0
#endif

__device__ __forceinline__ v8f wmma_bf16(v16bf a, v16bf b, v8f c) {
  return __builtin_amdgcn_wmma_f32_16x16x32_bf16(
      /*neg_a=*/false, a, /*neg_b=*/false, b,
      /*c_mod=*/(short)0, c, /*reuse_a=*/false, /*reuse_b=*/false);
}

__device__ __forceinline__ v16bf cat8(v8bf lo, v8bf hi) {
  return __builtin_shufflevector(lo, hi, 0,1,2,3,4,5,6,7,8,9,10,11,12,13,14,15);
}

#if USE_ASYNC_LDS
// async memory->LDS b128 copy: per-lane 16B, no VGPR round-trip (ASYNCcnt)
__device__ __forceinline__ void async_b128(const bf16* gsrc, bf16* ldst) {
  __builtin_amdgcn_global_load_async_to_lds_b128(
      (__attribute__((address_space(1))) v4i*)(v4i*)gsrc,
      (__attribute__((address_space(3))) v4i*)(v4i*)ldst, 0, 0);
}
#endif

// ---------------------------------------------------------------------------
// x (f32) -> bf16, 4 elements / thread
// ---------------------------------------------------------------------------
__global__ __launch_bounds__(256)
void convert_x_kernel(const float* __restrict__ x, bf16* __restrict__ xb) {
  int i = (blockIdx.x * 256 + threadIdx.x) * 4;
  float4 v = *(const float4*)(x + i);
  xb[i + 0] = (bf16)v.x;
  xb[i + 1] = (bf16)v.y;
  xb[i + 2] = (bf16)v.z;
  xb[i + 3] = (bf16)v.w;
}

// ---------------------------------------------------------------------------
// Transpose + convert weights: dst[n][k] = (bf16) src[k][n]
// z selects matrix: 0..2 -> Wqs/Wks/Wvs into WtQKV, 3 -> Wo into WtO
// ---------------------------------------------------------------------------
__global__ __launch_bounds__(256)
void transpose_w_kernel(const float* __restrict__ Wqs, const float* __restrict__ Wks,
                        const float* __restrict__ Wvs, const float* __restrict__ Wo,
                        bf16* __restrict__ WtQKV, bf16* __restrict__ WtO) {
  __shared__ float tile[32][33];
  const int z  = blockIdx.z;
  const float* src = (z == 0) ? Wqs : (z == 1) ? Wks : (z == 2) ? Wvs : Wo;
  bf16*        dst = (z < 3) ? (WtQKV + (size_t)z * DMODEL * DMODEL) : WtO;
  const int tx = threadIdx.x, ty = threadIdx.y;      // (32, 8)
  const int n0 = blockIdx.x * 32, k0 = blockIdx.y * 32;
#pragma unroll
  for (int j = 0; j < 32; j += 8)
    tile[ty + j][tx] = src[(size_t)(k0 + ty + j) * DMODEL + n0 + tx];
  __syncthreads();
#pragma unroll
  for (int j = 0; j < 32; j += 8)
    dst[(size_t)(n0 + ty + j) * DMODEL + k0 + tx] = (bf16)tile[tx][ty + j];
}

// ---------------------------------------------------------------------------
// bf16 WMMA GEMM: C[2048, N] = A[2048,1024] @ Bt[N,1024]^T
//   block tile 128x128, 8 waves (4 along M x 2 along N), wave tile 32x64.
//   Double-buffered LDS staging (one barrier per K-step); async mem->LDS
//   fills when available.
// mode 0: N=3072 ([q|k|v]); epilogue applies RoPE to q,k (bf16 out) and
//         writes v transposed per (head,dim) -> Vt[d_model][2048].
// mode 1: N=1024; plain f32 store to Cout.
// ---------------------------------------------------------------------------
__global__ __launch_bounds__(256)
void gemm_bf16_kernel(const bf16* __restrict__ A, const bf16* __restrict__ Bt,
                      int mode,
                      const float* __restrict__ cosT, const float* __restrict__ sinT,
                      const int* __restrict__ posIds,
                      bf16* __restrict__ Qout, bf16* __restrict__ Kout,
                      bf16* __restrict__ Vt, float* __restrict__ Cout) {
  __shared__ bf16 sA[2 * 128 * 32];
  __shared__ bf16 sB[2 * 128 * 32];
  const int t    = threadIdx.x;
  const int wave = t >> 5, lane = t & 31;
  const int wm   = wave & 3, wn = wave >> 2;          // 4 x 2 wave grid
  const int nl   = lane & 15, hi = lane >> 4;
  const int mbase = blockIdx.x * 128;
  const int nbase = blockIdx.y * 128;

  v8f acc[2][4];
#pragma unroll
  for (int mi = 0; mi < 2; ++mi)
#pragma unroll
    for (int ni = 0; ni < 4; ++ni) acc[mi][ni] = v8f{};

  const int srow  = t >> 1;          // 0..127
  const int skoff = (t & 1) * 16;    // 0 or 16 (bf16 elems)
  const bf16* gA = A  + (size_t)(mbase + srow) * DMODEL + skoff;
  const bf16* gB = Bt + (size_t)(nbase + srow) * DMODEL + skoff;
  const int lofs = srow * 32 + skoff;

  // Stage one 128x32 tile of A and B into LDS buffer `buf` for K offset k0.
  auto stage = [&](int buf, int k0) {
    bf16* la = sA + buf * (128 * 32) + lofs;
    bf16* lb = sB + buf * (128 * 32) + lofs;
#if USE_ASYNC_LDS
    async_b128(gA + k0,     la);
    async_b128(gA + k0 + 8, la + 8);
    async_b128(gB + k0,     lb);
    async_b128(gB + k0 + 8, lb + 8);
#else
    const uint4* ga4 = (const uint4*)(gA + k0);
    const uint4* gb4 = (const uint4*)(gB + k0);
    uint4 a0 = ga4[0], a1 = ga4[1];
    uint4 b0 = gb4[0], b1 = gb4[1];
    ((uint4*)la)[0] = a0; ((uint4*)la)[1] = a1;
    ((uint4*)lb)[0] = b0; ((uint4*)lb)[1] = b1;
    if (k0 + 32 < DMODEL) {   // pull next tile toward near caches
      __builtin_prefetch(gA + k0 + 32, 0, 3);
      __builtin_prefetch(gB + k0 + 32, 0, 3);
    }
#endif
  };

  stage(0, 0);
#if USE_ASYNC_LDS
  __asm__ volatile("s_wait_asynccnt 0x0" ::: "memory");
#endif
  __syncthreads();

  for (int k0 = 0; k0 < DMODEL; k0 += 32) {
    const int cur = (k0 >> 5) & 1;
    if (k0 + 32 < DMODEL) stage(cur ^ 1, k0 + 32);   // fill next buffer async

    const bf16* cA = sA + cur * (128 * 32);
    const bf16* cB = sB + cur * (128 * 32);
    // A fragments (contiguous K: lanes0-15 K{0..15}, lanes16-31 K{16..31})
    v16bf af[2];
#pragma unroll
    for (int mi = 0; mi < 2; ++mi) {
      const bf16* base = cA + (wm * 32 + mi * 16 + nl) * 32 + hi * 16;
      af[mi] = *(const v16bf*)base;
    }
    // B fragments (lane = N column, 16 contiguous K values)
    v16bf bfr[4];
#pragma unroll
    for (int ni = 0; ni < 4; ++ni)
      bfr[ni] = *(const v16bf*)(cB + (wn * 64 + ni * 16 + nl) * 32 + hi * 16);

#pragma unroll
    for (int mi = 0; mi < 2; ++mi)
#pragma unroll
      for (int ni = 0; ni < 4; ++ni)
        acc[mi][ni] = wmma_bf16(af[mi], bfr[ni], acc[mi][ni]);

#if USE_ASYNC_LDS
    __asm__ volatile("s_wait_asynccnt 0x0" ::: "memory");
#endif
    __syncthreads();
  }

  // Epilogue. C layout: element r of v8f -> row m = r + 8*hi, col n = nl.
  if (mode == 0) {
#pragma unroll
    for (int mi = 0; mi < 2; ++mi) {
#pragma unroll
      for (int ni = 0; ni < 4; ++ni) {
        const int n_g = nbase + wn * 64 + ni * 16 + nl;
        const int seg = n_g >> 10;        // 0=q 1=k 2=v
        const int col = n_g & 1023;
        const int d   = col & 63;         // head-dim (wave tiles are head-aligned)
#pragma unroll
        for (int r = 0; r < 8; ++r) {
          const int m_g = mbase + wm * 32 + mi * 16 + r + hi * 8;
          const float v = acc[mi][ni][r];
          if (seg == 2) {
            Vt[(size_t)col * SEQ + m_g] = (bf16)v;   // transposed per (h,d)
          } else {
            const int p = posIds[m_g];
            const float c  = cosT[p * HDIM + d];
            const float s  = sinT[p * HDIM + d];
            // rotate_half partner: tile ni^2 == d +/- 32, same reg/lane
            const float rot = (ni < 2) ? -acc[mi][ni + 2][r] : acc[mi][ni - 2][r];
            const float o = v * c + rot * s;
            if (seg == 0) Qout[(size_t)m_g * DMODEL + col] = (bf16)o;
            else          Kout[(size_t)m_g * DMODEL + col] = (bf16)o;
          }
        }
      }
    }
  } else {
    // plain f32 store
#pragma unroll
    for (int mi = 0; mi < 2; ++mi)
#pragma unroll
      for (int ni = 0; ni < 4; ++ni) {
        const int n_g = nbase + wn * 64 + ni * 16 + nl;
#pragma unroll
        for (int r = 0; r < 8; ++r) {
          const int m_g = mbase + wm * 32 + mi * 16 + r + hi * 8;
          Cout[(size_t)m_g * DMODEL + n_g] = acc[mi][ni][r];
        }
      }
  }
}

// ---------------------------------------------------------------------------
// Local sliding-window attention, flash-style, one wave per (head, 16 queries)
// Key tiles of 32. Tile 0 always processed (global columns 0..3 get the
// masked(-1e9)+boost(+1e9)=0 / in-window +1e9 semantics of the reference).
// ---------------------------------------------------------------------------
__global__ __launch_bounds__(128)
void attn_local_kernel(const bf16* __restrict__ Q, const bf16* __restrict__ K,
                       const bf16* __restrict__ Vt, float* __restrict__ Oatt) {
  __shared__ bf16 sP[4][16 * 32];
  const int t = threadIdx.x, wave = t >> 5, lane = t & 31;
  const int nl = lane & 15, hi = lane >> 4;
  const int h  = blockIdx.x >> 5;                 // 16 heads
  const int qb = (blockIdx.x & 31) * 4 + wave;    // 0..127
  const int qbase = qb * 16;
  bf16* myP = sP[wave];
  const float SCALE = 0.125f, NEGV = -1e9f, BOOST = 1e9f;

  // Q fragments (A layout), head-dim 64 -> two K=32 steps
  const bf16* qrow = Q + (size_t)(qbase + nl) * DMODEL + h * HDIM;
  v16bf aq[2];
#pragma unroll
  for (int ks = 0; ks < 2; ++ks) {
    const int c0 = ks * 32 + hi * 16;
    aq[ks] = *(const v16bf*)(qrow + c0);
  }

  float mrun[8], lrun[8];
  v8f accO[4];
#pragma unroll
  for (int r = 0; r < 8; ++r) { mrun[r] = -3.0e38f; lrun[r] = 0.f; }
#pragma unroll
  for (int j = 0; j < 4; ++j) accO[j] = v8f{};

  auto proc = [&](int kt) {
    const int kbase = kt * 32;
    // S = Q @ K^T for 32 keys -> two 16-key C tiles
    v8f s0 = v8f{}, s1 = v8f{};
    const bf16* kr0 = K + (size_t)(kbase + nl) * DMODEL + h * HDIM;
    const bf16* kr1 = K + (size_t)(kbase + 16 + nl) * DMODEL + h * HDIM;
#pragma unroll
    for (int ks = 0; ks < 2; ++ks) {
      v16bf b0 = *(const v16bf*)(kr0 + ks * 32 + hi * 16);
      v16bf b1 = *(const v16bf*)(kr1 + ks * 32 + hi * 16);
      s0 = wmma_bf16(aq[ks], b0, s0);
      s1 = wmma_bf16(aq[ks], b1, s1);
    }
    float p0[8], p1[8], alpha[8];
#pragma unroll
    for (int r = 0; r < 8; ++r) {
      const int i  = qbase + r + hi * 8;
      const int j0 = kbase + nl, j1 = kbase + 16 + nl;
      float v0 = s0[r] * SCALE, v1 = s1[r] * SCALE;
      if (!((j0 <= i) && (i - j0 <= RADIUS))) v0 = NEGV;
      if (!((j1 <= i) && (i - j1 <= RADIUS))) v1 = NEGV;
      if (j0 < 4) v0 += BOOST;
      if (j1 < 4) v1 += BOOST;
      p0[r] = v0; p1[r] = v1;
    }
    // online softmax: per-row max/sum across 16 lanes of each half-wave
#pragma unroll
    for (int r = 0; r < 8; ++r) {
      float tm = fmaxf(p0[r], p1[r]);
      tm = fmaxf(tm, __shfl_xor(tm, 1));
      tm = fmaxf(tm, __shfl_xor(tm, 2));
      tm = fmaxf(tm, __shfl_xor(tm, 4));
      tm = fmaxf(tm, __shfl_xor(tm, 8));
      const float mn = fmaxf(mrun[r], tm);
      alpha[r] = __expf(mrun[r] - mn);
      mrun[r]  = mn;
      p0[r] = __expf(p0[r] - mn);
      p1[r] = __expf(p1[r] - mn);
      float rs = p0[r] + p1[r];
      rs += __shfl_xor(rs, 1);
      rs += __shfl_xor(rs, 2);
      rs += __shfl_xor(rs, 4);
      rs += __shfl_xor(rs, 8);
      lrun[r] = lrun[r] * alpha[r] + rs;
    }
#pragma unroll
    for (int j = 0; j < 4; ++j)
#pragma unroll
      for (int r = 0; r < 8; ++r) accO[j][r] *= alpha[r];
    // re-layout P (C layout -> A layout) through LDS
#pragma unroll
    for (int r = 0; r < 8; ++r) {
      myP[(r + hi * 8) * 32 + nl]      = (bf16)p0[r];
      myP[(r + hi * 8) * 32 + 16 + nl] = (bf16)p1[r];
    }
    __asm__ volatile("s_wait_dscnt 0x0" ::: "memory");
    const bf16* pr = myP + nl * 32 + hi * 16;
    v16bf pa = *(const v16bf*)pr;
    // O += P @ V  (V transposed: per lane, 16 contiguous keys)
#pragma unroll
    for (int j = 0; j < 4; ++j) {
      const bf16* vr = Vt + (size_t)(h * HDIM + j * 16 + nl) * SEQ + kbase + hi * 16;
      accO[j] = wmma_bf16(pa, *(const v16bf*)vr, accO[j]);
    }
  };

  proc(0);  // always: holds the boosted global columns 0..3
  int kt0 = (qbase > RADIUS) ? ((qbase - RADIUS) >> 5) : 1;
  if (kt0 < 1) kt0 = 1;
  const int ktLast = (qbase + 15) >> 5;
  for (int kt = kt0; kt <= ktLast; ++kt) proc(kt);

#pragma unroll
  for (int j = 0; j < 4; ++j)
#pragma unroll
    for (int r = 0; r < 8; ++r) {
      const int m_g = qbase + r + hi * 8;
      Oatt[(size_t)m_g * DMODEL + h * HDIM + j * 16 + nl] = accO[j][r] / lrun[r];
    }
}

// ---------------------------------------------------------------------------
// kg/vg only needed at the 4 global rows: [4,1024] @ [1024,1024], f32 exact
// ---------------------------------------------------------------------------
__global__ __launch_bounds__(256)
void kgvg_kernel(const float* __restrict__ x, const float* __restrict__ Wkg,
                 const float* __restrict__ Wvg, float* __restrict__ kg4,
                 float* __restrict__ vg4) {
  const int idx   = blockIdx.x * 256 + threadIdx.x;   // 8192
  const int which = idx >> 12;
  const int row   = (idx >> 10) & 3;
  const int col   = idx & 1023;
  const float* W  = which ? Wvg : Wkg;
  const float* xr = x + (size_t)row * DMODEL;
  float acc = 0.f;
  for (int k = 0; k < DMODEL; ++k) acc += xr[k] * W[(size_t)k * DMODEL + col];
  (which ? vg4 : kg4)[row * DMODEL + col] = acc;
}

// ---------------------------------------------------------------------------
// Global attention (4 keys) + add local output, emit bf16 for final GEMM
// ---------------------------------------------------------------------------
__global__ __launch_bounds__(256)
void attn_global_kernel(const bf16* __restrict__ Q, const float* __restrict__ kg4,
                        const float* __restrict__ vg4, const float* __restrict__ Oatt,
                        bf16* __restrict__ ab) {
  const int idx = blockIdx.x * 256 + threadIdx.x;     // 32768
  const int h = idx & 15, s = idx >> 4;
  const bf16* q = Q + (size_t)s * DMODEL + h * HDIM;
  float sc[4] = {0.f, 0.f, 0.f, 0.f};
  for (int d = 0; d < HDIM; ++d) {
    const float qv = (float)q[d];
#pragma unroll
    for (int tt = 0; tt < 4; ++tt) sc[tt] += qv * kg4[tt * DMODEL + h * HDIM + d];
  }
  float m = -3.0e38f;
#pragma unroll
  for (int tt = 0; tt < 4; ++tt) { sc[tt] *= 0.125f; m = fmaxf(m, sc[tt]); }
  float p[4], sum = 0.f;
#pragma unroll
  for (int tt = 0; tt < 4; ++tt) { p[tt] = __expf(sc[tt] - m); sum += p[tt]; }
  const float inv = 1.f / sum;
  for (int d = 0; d < HDIM; ++d) {
    float o = 0.f;
#pragma unroll
    for (int tt = 0; tt < 4; ++tt) o += p[tt] * vg4[tt * DMODEL + h * HDIM + d];
    const size_t off = (size_t)s * DMODEL + h * HDIM + d;
    ab[off] = (bf16)(o * inv + Oatt[off]);
  }
}

// ---------------------------------------------------------------------------
extern "C" void kernel_launch(void* const* d_in, const int* in_sizes, int n_in,
                              void* d_out, int out_size, void* d_ws, size_t ws_size,
                              hipStream_t stream) {
  (void)in_sizes; (void)n_in; (void)out_size; (void)ws_size;
  const float* x    = (const float*)d_in[0];
  const float* cosT = (const float*)d_in[1];
  const float* sinT = (const float*)d_in[2];
  const int*   pos  = (const int*)d_in[3];
  const float* Wqs  = (const float*)d_in[4];
  const float* Wks  = (const float*)d_in[5];
  const float* Wvs  = (const float*)d_in[6];
  // d_in[7] = Wqg (unused by the reference forward math)
  const float* Wkg  = (const float*)d_in[8];
  const float* Wvg  = (const float*)d_in[9];
  const float* Wo   = (const float*)d_in[10];
  float* out = (float*)d_out;

  char* ws = (char*)d_ws;
  bf16*  xb    = (bf16*)(ws);                       //  4 MiB
  bf16*  WtQKV = (bf16*)(ws + (4u  << 20));         //  6 MiB
  bf16*  WtO   = (bf16*)(ws + (10u << 20));         //  2 MiB
  bf16*  Qb    = (bf16*)(ws + (12u << 20));         //  4 MiB
  bf16*  Kb    = (bf16*)(ws + (16u << 20));         //  4 MiB
  bf16*  Vt    = (bf16*)(ws + (20u << 20));         //  4 MiB
  float* Oatt  = (float*)(ws + (24u << 20));        //  8 MiB
  bf16*  ab    = (bf16*)(ws + (32u << 20));         //  4 MiB
  float* kg4   = (float*)(ws + (36u << 20));        // 16 KiB
  float* vg4   = (float*)(ws + (36u << 20) + 16384);

  convert_x_kernel<<<2048, 256, 0, stream>>>(x, xb);
  transpose_w_kernel<<<dim3(32, 32, 4), dim3(32, 8), 0, stream>>>(
      Wqs, Wks, Wvs, Wo, WtQKV, WtO);
  gemm_bf16_kernel<<<dim3(16, 24), 256, 0, stream>>>(
      xb, WtQKV, 0, cosT, sinT, pos, Qb, Kb, Vt, nullptr);
  kgvg_kernel<<<32, 256, 0, stream>>>(x, Wkg, Wvg, kg4, vg4);
  attn_local_kernel<<<512, 128, 0, stream>>>(Qb, Kb, Vt, Oatt);
  attn_global_kernel<<<128, 256, 0, stream>>>(Qb, kg4, vg4, Oatt, ab);
  gemm_bf16_kernel<<<dim3(16, 8), 256, 0, stream>>>(
      ab, WtO, 1, nullptr, nullptr, nullptr, nullptr, nullptr, nullptr, out);
}
